// HivemindAgent_91139206021334
// MI455X (gfx1250) — matrix operand order repaired and checked by reading, not compiled, not run on hardware
//
#include <hip/hip_runtime.h>

#define B_    64
#define S_    512
#define D_    512
#define H_    8
#define L_    4
#define FF_   2048
#define HOUT_ 64

typedef __bf16 bf16;
typedef __attribute__((ext_vector_type(16))) bf16     v16bf;
typedef __attribute__((ext_vector_type(8)))  float    v8f;
typedef __attribute__((ext_vector_type(4)))  unsigned u32x4;
typedef __attribute__((ext_vector_type(8)))  unsigned u32x8;

static __device__ __forceinline__ bf16 f2b(float f) { return (bf16)f; }

// ---------------------------------------------------------------------------
// Embedding gather: h[b,s,:] = emb[x[b,s],:]
// ---------------------------------------------------------------------------
__global__ void embed_kernel(const int* __restrict__ x, const float* __restrict__ emb,
                             float* __restrict__ h) {
  int row = blockIdx.x;                 // b*S + s
  int tok = x[row];
  int t = threadIdx.x;                  // 128 threads * float4 = 512 floats
  const float4* src = (const float4*)(emb + (size_t)tok * D_);
  float4* dst = (float4*)(h + (size_t)row * D_);
  dst[t] = src[t];
}

// ---------------------------------------------------------------------------
// Tiled WMMA GEMM: C[M,N] = A[M,K] @ W[K,N] + bias[N]  (optional ReLU)
// A: f32 or bf16 (ABF16). C: f32 or bf16 (OUTBF16). W,bias: f32, row stride ldw.
// Block 256 threads (8 waves, 4x2 wave grid), tile 128x128x32, bf16 WMMA f32-acc.
// Double-buffered LDS (stage k+1 while computing k, one barrier per step).
// LDS row stride 40 halfwords (bank-conflict padding).
// bf16-A path stages the A tile with the Tensor Data Mover (TDM): hardware
// padding (pad_interval=16 dwords, pad_amount=4 dwords) reproduces the
// 40-halfword stride; wave 0 issues, s_wait_tensorcnt 0 before the barrier.
// ---------------------------------------------------------------------------
template<bool ABF16, bool RELU, bool OUTBF16>
__global__ void gemm_wmma(const void* __restrict__ A_, const float* __restrict__ W,
                          const float* __restrict__ bias, void* __restrict__ C_,
                          int M, int N, int K, int ldw) {
  constexpr int LSTR = 40;
  __shared__ bf16 As[2][128 * LSTR];
  __shared__ bf16 Bs[2][128 * LSTR];

  const int t = threadIdx.x;
  const int lane = t & 31, w = t >> 5;
  const int wm = w >> 1, wn = w & 1;          // wave grid 4 (m) x 2 (n)
  const int half = lane >> 4, l16 = lane & 15;
  const int bn = blockIdx.x, bm = blockIdx.y;
  const int KT = K >> 5;                      // K / 32 steps

  const v8f vzero = {0.f,0.f,0.f,0.f,0.f,0.f,0.f,0.f};
  v8f acc[2][4];
  #pragma unroll
  for (int i = 0; i < 2; ++i)
    #pragma unroll
    for (int j = 0; j < 4; ++j) acc[i][j] = vzero;

  // ---- tile stagers -------------------------------------------------------
  auto stageA_f32 = [&](int buf, int k0) {
    const float* A = (const float*)A_;
    #pragma unroll
    for (int i = 0; i < 4; ++i) {
      int linear = t + i * 256;               // 1024 float4 chunks
      int row = linear >> 3;
      int k4 = (linear & 7) * 4;
      int grow = bm * 128 + row;
      float4 v = {0.f,0.f,0.f,0.f};
      if (grow < M) {
        v = *(const float4*)(A + (size_t)grow * K + k0 + k4);
        if (i == 0 && k0 + 64 < K)            // prefetch 2 tiles ahead
          __builtin_prefetch(A + (size_t)grow * K + k0 + 64 + k4, 0, 0);
      }
      bf16* d = &As[buf][row * LSTR + k4];
      d[0] = f2b(v.x); d[1] = f2b(v.y); d[2] = f2b(v.z); d[3] = f2b(v.w);
    }
  };

  auto issueA_tdm = [&](int buf, int k0) {    // bf16 A tile via Tensor Data Mover
    if (t < 32) {                             // one wave issues (TDM ignores EXEC)
      const bf16* gptr = (const bf16*)A_ + (size_t)(bm * 128) * K + k0;
      unsigned long long ga = (unsigned long long)(uintptr_t)gptr;
      unsigned lds = (unsigned)(uintptr_t)&As[buf][0];   // low 32 bits = LDS offset
      u32x4 g0;
      g0.x = 1u;                                          // count=1, user mode
      g0.y = lds;                                         // lds_addr
      g0.z = (unsigned)ga;                                // global_addr[31:0]
      g0.w = (unsigned)((ga >> 32) & 0x1FFFFFFu) | (2u << 30);  // addr hi | type=2
      u32x8 g1;
      g1.s0 = (1u << 16)        // data_size = 2 bytes
            | (1u << 20)        // pad_enable
            | (3u << 22)        // pad_interval: 16 dwords (one 32-elem bf16 row)
            | (3u << 25);       // pad_amount: 4 dwords (16B -> stride 40 bf16)
      g1.s1 = ((unsigned)K & 0xFFFFu) << 16;                       // tensor_dim0 lo
      g1.s2 = ((unsigned)K >> 16) | (((unsigned)M & 0xFFFFu) << 16); // dim0 hi|dim1 lo
      g1.s3 = ((unsigned)M >> 16) | (32u << 16);                   // dim1 hi|tile_dim0=32
      g1.s4 = 128u;                                                // tile_dim1=128
      g1.s5 = (unsigned)K;                                         // dim0_stride lo
      g1.s6 = 0u;                                                  // stride hi / dim1_stride lo
      g1.s7 = 0u;
      asm volatile("tensor_load_to_lds %0, %1" :: "s"(g0), "s"(g1) : "memory");
    }
  };

  auto stageB = [&](int buf, int k0) {        // W tile (32K x 128N) transposed
    #pragma unroll
    for (int i = 0; i < 4; ++i) {
      int linear = t + i * 256;
      int kk = linear >> 5;
      int n4 = (linear & 31) * 4;
      float4 v = *(const float4*)(W + (size_t)(k0 + kk) * ldw + bn * 128 + n4);
      if (i == 0 && k0 + 64 < K)
        __builtin_prefetch(W + (size_t)(k0 + 64 + kk) * ldw + bn * 128 + n4, 0, 0);
      Bs[buf][(n4 + 0) * LSTR + kk] = f2b(v.x);
      Bs[buf][(n4 + 1) * LSTR + kk] = f2b(v.y);
      Bs[buf][(n4 + 2) * LSTR + kk] = f2b(v.z);
      Bs[buf][(n4 + 3) * LSTR + kk] = f2b(v.w);
    }
  };

  // ---- prologue: stage tile 0 ----
  if constexpr (ABF16) issueA_tdm(0, 0); else stageA_f32(0, 0);
  stageB(0, 0);
  if constexpr (ABF16) { if (t < 32) __builtin_amdgcn_s_wait_tensorcnt(0); }
  __syncthreads();

  // ---- main loop: compute cur, stage nxt ----
  for (int kt = 0; kt < KT; ++kt) {
    const int cur = kt & 1, nxt = cur ^ 1;
    if (kt + 1 < KT) {
      if constexpr (ABF16) issueA_tdm(nxt, (kt + 1) * 32);
      else                 stageA_f32(nxt, (kt + 1) * 32);
      stageB(nxt, (kt + 1) * 32);
    }

    // fragments in contiguous k order (k = 16*half + e), same for A and B
    v16bf af[2], bfv[4];
    const bf16* Ab = As[cur];
    const bf16* Bb = Bs[cur];
    #pragma unroll
    for (int mf = 0; mf < 2; ++mf) {
      const bf16* p = &Ab[(wm * 32 + mf * 16 + l16) * LSTR + 16 * half];
      #pragma unroll
      for (int e = 0; e < 8; ++e) { af[mf][e] = p[e]; af[mf][e + 8] = p[e + 8]; }
    }
    #pragma unroll
    for (int nf = 0; nf < 4; ++nf) {
      const bf16* p = &Bb[(wn * 64 + nf * 16 + l16) * LSTR + 16 * half];
      #pragma unroll
      for (int e = 0; e < 16; ++e) bfv[nf][e] = p[e];
    }
    #pragma unroll
    for (int mf = 0; mf < 2; ++mf)
      #pragma unroll
      for (int nf = 0; nf < 4; ++nf)
        acc[mf][nf] = __builtin_amdgcn_wmma_f32_16x16x32_bf16(
            false, af[mf], false, bfv[nf], (short)0, acc[mf][nf], false, false);

    if constexpr (ABF16) {
      if (t < 32 && kt + 1 < KT) __builtin_amdgcn_s_wait_tensorcnt(0);
    }
    __syncthreads();
  }

  // ---- epilogue: bias (+ReLU), store f32 or bf16 ----
  #pragma unroll
  for (int mf = 0; mf < 2; ++mf)
    #pragma unroll
    for (int nf = 0; nf < 4; ++nf)
      #pragma unroll
      for (int r = 0; r < 8; ++r) {
        int row = bm * 128 + wm * 32 + mf * 16 + r + 8 * half;
        int col = bn * 128 + wn * 64 + nf * 16 + l16;
        if (row < M) {
          float v = acc[mf][nf][r] + bias[col];
          if (RELU) v = fmaxf(v, 0.f);
          if (OUTBF16) ((bf16*)C_)[(size_t)row * N + col] = f2b(v);
          else         ((float*)C_)[(size_t)row * N + col] = v;
        }
      }
}

// ---------------------------------------------------------------------------
// Causal self-attention for one (b, head, 16-q-row tile).
// qkv: [B,S,3D] f32 (q|k|v). Output aout: [B,S,D] f32.
// Block 128 threads (4 waves). Scores held in LDS f32; softmax over causal len.
// ---------------------------------------------------------------------------
__global__ void attn_kernel(const float* __restrict__ qkv, float* __restrict__ aout) {
  constexpr int SCS = 520;   // score row stride (f32)
  constexpr int VSS = 40;    // V-tile row stride (bf16)
  __shared__ float sc[16 * SCS];
  __shared__ bf16  Vs[64 * VSS];
  __shared__ float red[16 * 8];

  const int qt = blockIdx.x, hh = blockIdx.y, b = blockIdx.z;
  const int qbase = qt * 16;
  const int limit = qbase + 16;             // kv needed by this tile (causal)
  const int limitR = (limit + 31) & ~31;    // round up to 32 for P@V steps
  const int t = threadIdx.x;
  const int lane = t & 31, w = t >> 5;
  const int half = lane >> 4, l16 = lane & 15;
  const v8f vzero = {0.f,0.f,0.f,0.f,0.f,0.f,0.f,0.f};

  // ---- phase 1: scores = (Q K^T) * 1/sqrt(dh) ----
  v16bf qf[2];
  {
    const float* qp = qkv + (size_t)(b * S_ + qbase + l16) * (3 * D_) + hh * 64 + 16 * half;
    #pragma unroll
    for (int c = 0; c < 2; ++c)
      #pragma unroll
      for (int e = 0; e < 8; ++e) {
        qf[c][e]     = f2b(qp[32 * c + e]);
        qf[c][e + 8] = f2b(qp[32 * c + 8 + e]);
      }
  }
  for (int fi = w; fi * 16 < limit; fi += 4) {   // round-robin kv frags to waves
    int kv0 = fi * 16;
    v8f s = vzero;
    #pragma unroll
    for (int c = 0; c < 2; ++c) {
      v16bf kf;   // B-frag: N=kv row (contiguous d reads -> no transpose needed)
      const float* kp = qkv + (size_t)(b * S_ + kv0 + l16) * (3 * D_) + D_ + hh * 64
                        + 16 * half + 32 * c;
      #pragma unroll
      for (int e = 0; e < 16; ++e) kf[e] = f2b(kp[e]);
      s = __builtin_amdgcn_wmma_f32_16x16x32_bf16(false, qf[c], false, kf, (short)0,
                                                  s, false, false);
    }
    #pragma unroll
    for (int r = 0; r < 8; ++r)
      sc[(r + 8 * half) * SCS + kv0 + l16] = s[r] * 0.125f;   // 1/sqrt(64)
  }
  __syncthreads();

  // ---- phase 2: causal softmax, 8 threads per row ----
  {
    int row = t >> 3, seg = t & 7;
    int len = qbase + row + 1;                   // causal valid length
    float* srow = &sc[row * SCS];
    float m = -1e30f;
    for (int c = seg; c < len; c += 8) m = fmaxf(m, srow[c]);
    red[row * 8 + seg] = m;
    __syncthreads();
    float rm = -1e30f;
    #pragma unroll
    for (int j = 0; j < 8; ++j) rm = fmaxf(rm, red[row * 8 + j]);
    __syncthreads();
    float sum = 0.f;
    for (int c = seg; c < len; c += 8) { float e = __expf(srow[c] - rm); srow[c] = e; sum += e; }
    red[row * 8 + seg] = sum;
    __syncthreads();
    float tot = 0.f;
    #pragma unroll
    for (int j = 0; j < 8; ++j) tot += red[row * 8 + j];
    float inv = 1.f / tot;
    for (int c = seg; c < len; c += 8) srow[c] *= inv;
    int c0 = len + (((seg - len) % 8) + 8) % 8;  // zero masked tail for P@V
    for (int c = c0; c < limitR; c += 8) srow[c] = 0.f;
  }
  __syncthreads();

  // ---- phase 3: O = P @ V, each wave owns a 16-wide d slice ----
  v8f o = vzero;
  const int nsteps = limitR >> 5;
  for (int st = 0; st < nsteps; ++st) {
    int kv0 = st * 32;
    #pragma unroll
    for (int i = 0; i < 16; ++i) {               // stage V (32 kv x 64 d) transposed
      int linear = t + i * 128;
      int kvr = linear >> 6, d = linear & 63;
      Vs[d * VSS + kvr] =
          f2b(qkv[(size_t)(b * S_ + kv0 + kvr) * (3 * D_) + 2 * D_ + hh * 64 + d]);
    }
    __syncthreads();
    v16bf pf, vf;
    const float* pp = &sc[l16 * SCS + kv0 + 16 * half];
    #pragma unroll
    for (int e = 0; e < 8; ++e) { pf[e] = f2b(pp[e]); pf[e + 8] = f2b(pp[e + 8]); }
    const bf16* vp = &Vs[(w * 16 + l16) * VSS + 16 * half];
    #pragma unroll
    for (int e = 0; e < 16; ++e) vf[e] = vp[e];
    o = __builtin_amdgcn_wmma_f32_16x16x32_bf16(false, pf, false, vf, (short)0,
                                                o, false, false);
    __syncthreads();
  }
  #pragma unroll
  for (int r = 0; r < 8; ++r) {
    int q = qbase + r + 8 * half;
    aout[(size_t)(b * S_ + q) * D_ + hh * 64 + w * 16 + l16] = o[r];
  }
}

// ---------------------------------------------------------------------------
// h = LayerNorm(h + add) * g + beta  (row = b*S+s, D=512, 256 threads)
// ---------------------------------------------------------------------------
__global__ void ln_kernel(float* __restrict__ h, const float* __restrict__ add,
                          const float* __restrict__ g, const float* __restrict__ beta) {
  __shared__ float s1[256], s2[256];
  int row = blockIdx.x, t = threadIdx.x;
  float* hr = h + (size_t)row * D_;
  const float* ar = add + (size_t)row * D_;
  float v0 = hr[t] + ar[t];
  float v1 = hr[t + 256] + ar[t + 256];
  s1[t] = v0 + v1;
  s2[t] = v0 * v0 + v1 * v1;
  __syncthreads();
  for (int off = 128; off > 0; off >>= 1) {
    if (t < off) { s1[t] += s1[t + off]; s2[t] += s2[t + off]; }
    __syncthreads();
  }
  float mu = s1[0] * (1.f / D_);
  float var = s2[0] * (1.f / D_) - mu * mu;
  float inv = rsqrtf(var + 1e-5f);
  hr[t]       = (v0 - mu) * inv * g[t] + beta[t];
  hr[t + 256] = (v1 - mu) * inv * g[t + 256] + beta[t + 256];
}

// ctx = context @ ctx_w + ctx_b   (64 x 2) @ (2 x 512)
__global__ void ctx_kernel(const float* __restrict__ ctx, const float* __restrict__ w,
                           const float* __restrict__ bias, float* __restrict__ out) {
  int b = blockIdx.x, t = threadIdx.x;
  float c0 = ctx[b * 2], c1 = ctx[b * 2 + 1];
  for (int d = t; d < D_; d += 256)
    out[b * D_ + d] = c0 * w[d] + c1 * w[D_ + d] + bias[d];
}

// pred/act = einsum('bd,bdo', last, W[id]) + b[id]
__global__ void spec_kernel(const float* __restrict__ last, const int* __restrict__ ids,
                            const float* __restrict__ pw, const float* __restrict__ pb,
                            const float* __restrict__ aw, const float* __restrict__ ab,
                            float* __restrict__ out) {
  int b = blockIdx.x, o = threadIdx.x;
  int id = ids[b];
  float ap = pb[id * HOUT_ + o];
  float aa = ab[id * HOUT_ + o];
  const float* lr = last + b * D_;
  const float* pwp = pw + (size_t)id * D_ * HOUT_ + o;
  const float* awp = aw + (size_t)id * D_ * HOUT_ + o;
  for (int d = 0; d < D_; ++d) {
    float l = lr[d];
    ap += l * pwp[d * HOUT_];
    aa += l * awp[d * HOUT_];
  }
  out[b * HOUT_ + o] = ap;
  out[B_ * HOUT_ + b * HOUT_ + o] = aa;
}

// ---------------------------------------------------------------------------
extern "C" void kernel_launch(void* const* d_in, const int* in_sizes, int n_in,
                              void* d_out, int out_size, void* d_ws, size_t ws_size,
                              hipStream_t stream) {
  (void)in_sizes; (void)n_in; (void)out_size; (void)ws_size;
  const int*   x          = (const int*)  d_in[0];
  const float* context    = (const float*)d_in[1];
  const int*   ids        = (const int*)  d_in[2];
  const float* emb        = (const float*)d_in[3];
  const float* enc_in_w   = (const float*)d_in[4];
  const float* enc_in_b   = (const float*)d_in[5];
  const float* enc_out_w  = (const float*)d_in[6];
  const float* enc_out_b  = (const float*)d_in[7];
  const float* ln1_g      = (const float*)d_in[8];
  const float* ln1_b      = (const float*)d_in[9];
  const float* lin1_w     = (const float*)d_in[10];
  const float* lin1_b     = (const float*)d_in[11];
  const float* lin2_w     = (const float*)d_in[12];
  const float* lin2_b     = (const float*)d_in[13];
  const float* ln2_g      = (const float*)d_in[14];
  const float* ln2_b      = (const float*)d_in[15];
  const float* ctx_w      = (const float*)d_in[16];
  const float* ctx_b      = (const float*)d_in[17];
  const float* ca_in_w    = (const float*)d_in[18];
  const float* ca_in_b    = (const float*)d_in[19];
  const float* ca_out_w   = (const float*)d_in[20];
  const float* ca_out_b   = (const float*)d_in[21];
  const float* spec_pred_w= (const float*)d_in[22];
  const float* spec_pred_b= (const float*)d_in[23];
  const float* spec_act_w = (const float*)d_in[24];
  const float* spec_act_b = (const float*)d_in[25];

  // workspace layout (~336 MB):
  //   h:    [B,S,D]  f32   64 MB
  //   qkv:  [B,S,3D] f32  192 MB   (reused: out-proj tmp f32 / FFN mid bf16)
  //   aout: [B,S,D]  f32   64 MB
  //   ctxbuf/vca/last: 3 x [B,D] f32
  char* ws = (char*)d_ws;
  const size_t HB = (size_t)B_ * S_ * D_ * sizeof(float);
  float* h      = (float*)ws;
  void*  qkv    = (void*)(ws + HB);
  float* aout   = (float*)(ws + HB + (size_t)B_ * S_ * 3 * D_ * sizeof(float));
  float* ctxbuf = (float*)((char*)aout + HB);
  float* vca    = ctxbuf + B_ * D_;
  float* lastb  = vca + B_ * D_;

  const int M = B_ * S_;   // 32768

  embed_kernel<<<M, 128, 0, stream>>>(x, emb, h);

  for (int i = 0; i < L_; ++i) {
    // QKV projection: [M,512] @ [512,1536]
    gemm_wmma<false,false,false><<<dim3(12, 256), 256, 0, stream>>>(
        h, enc_in_w + (size_t)i * D_ * 3 * D_, enc_in_b + i * 3 * D_,
        qkv, M, 3 * D_, D_, 3 * D_);
    // causal self-attention
    attn_kernel<<<dim3(S_ / 16, H_, B_), 128, 0, stream>>>((const float*)qkv, aout);
    // output projection -> reuse qkv region as f32 tmp
    gemm_wmma<false,false,false><<<dim3(4, 256), 256, 0, stream>>>(
        aout, enc_out_w + (size_t)i * D_ * D_, enc_out_b + i * D_,
        qkv, M, D_, D_, D_);
    ln_kernel<<<M, 256, 0, stream>>>(h, (const float*)qkv, ln1_g + i * D_, ln1_b + i * D_);
    // FFN up + ReLU -> bf16 mid (reuses qkv region)
    gemm_wmma<false,true,true><<<dim3(16, 256), 256, 0, stream>>>(
        h, lin1_w + (size_t)i * D_ * FF_, lin1_b + i * FF_,
        qkv, M, FF_, D_, FF_);
    // FFN down (bf16 A staged via TDM tensor_load_to_lds)
    gemm_wmma<true,false,false><<<dim3(4, 256), 256, 0, stream>>>(
        qkv, lin2_w + (size_t)i * FF_ * D_, lin2_b + i * D_,
        aout, M, D_, FF_, D_);
    ln_kernel<<<M, 256, 0, stream>>>(h, aout, ln2_g + i * D_, ln2_b + i * D_);
  }

  // cross-attn with kv-len 1: softmax==1 => latent[:, -1] = (ctx@Wv+bv)@Wo+bo
  ctx_kernel<<<B_, 256, 0, stream>>>(context, ctx_w, ctx_b, ctxbuf);
  gemm_wmma<false,false,false><<<dim3(4, 1), 256, 0, stream>>>(
      ctxbuf, ca_in_w + 2 * D_, ca_in_b + 2 * D_, vca, B_, D_, D_, 3 * D_);
  gemm_wmma<false,false,false><<<dim3(4, 1), 256, 0, stream>>>(
      vca, ca_out_w, ca_out_b, lastb, B_, D_, D_, D_);

  spec_kernel<<<B_, HOUT_, 0, stream>>>(lastb, ids, spec_pred_w, spec_pred_b,
                                        spec_act_w, spec_act_b, (float*)d_out);
}
